// HYVideoDiffusionTransformer_8873402434118
// MI455X (gfx1250) — compile-verified, matrix operand order checked by reading, not run on hardware
//
#include <hip/hip_runtime.h>
#include <hip/hip_bf16.h>
#include <math.h>

typedef __bf16 bf16;
typedef __attribute__((ext_vector_type(16))) __bf16 v16bf;
typedef __attribute__((ext_vector_type(8)))  __bf16 v8bf;
typedef __attribute__((ext_vector_type(8)))  float  v8f;

#define HID   3072
#define HEADS 24
#define HD    128
#define MLPD  12288
#define QKVN  9216
#define L_IMG 2048
#define L_TXT 256
#define L_TOT 2304
#define EPSF  1e-6f

__device__ __forceinline__ v8f vzero8() {
    v8f z;
#pragma unroll
    for (int i = 0; i < 8; ++i) z[i] = 0.f;
    return z;
}

// Load a 16x32 (MxK) 16-bit WMMA operand fragment row for this lane.
// CDNA5 layout: lane = khalf*16 + m; VGPR0-3 hold K = khalf*8+0..7 (packed pairs),
// VGPR4-7 hold K = 16+khalf*8+0..7.
__device__ __forceinline__ v16bf frag_ld(const bf16* rowptr, int khalf) {
    union { v16bf v; v8bf h[2]; } u;
    u.h[0] = *(const v8bf*)(rowptr + khalf * 8);
    u.h[1] = *(const v8bf*)(rowptr + 16 + khalf * 8);
    return u.v;
}

__device__ __forceinline__ v8f wmma_bf16(v16bf a, v16bf b, v8f c) {
    return __builtin_amdgcn_wmma_f32_16x16x32_bf16(false, a, false, b, (short)0, c,
                                                   false, false);
}

// ---------------------------------------------------------------------------
// silu(vec) -> va
__global__ __launch_bounds__(256) void silu_kernel(const float* __restrict__ v,
                                                   float* __restrict__ o) {
    int i = blockIdx.x * 256 + threadIdx.x;
    if (i < HID) { float x = v[i]; o[i] = x / (1.f + __expf(-x)); }
}

// modulation GEMV: [1,3072] @ [3072,18432] + b, both streams
__global__ __launch_bounds__(256) void mod_gemv(const float* __restrict__ va,
                                                const float* __restrict__ Wi,
                                                const float* __restrict__ bi,
                                                const float* __restrict__ Wt,
                                                const float* __restrict__ bt,
                                                float* __restrict__ mi,
                                                float* __restrict__ mt) {
    int j = blockIdx.x * 256 + threadIdx.x;     // 0..36863
    const float *W, *b; float* o; int jj;
    if (j < 6 * HID) { W = Wi; b = bi; o = mi; jj = j; }
    else             { W = Wt; b = bt; o = mt; jj = j - 6 * HID; }
    float acc = b[jj];
    for (int k = 0; k < HID; ++k) acc += va[k] * W[(size_t)k * (6 * HID) + jj];
    o[jj] = acc;
}

// fp32 [K][N] -> bf16 transposed [N][K]
__global__ __launch_bounds__(256) void w_to_bf16_t(const float* __restrict__ W,
                                                   bf16* __restrict__ Wt,
                                                   int Kd, int Nd) {
    __shared__ bf16 t[32][33];
    int n0 = blockIdx.x * 32, k0 = blockIdx.y * 32;
    int c = threadIdx.x & 31, r0 = threadIdx.x >> 5;
#pragma unroll
    for (int rr = 0; rr < 32; rr += 8)
        t[r0 + rr][c] = (bf16)W[(size_t)(k0 + r0 + rr) * Nd + n0 + c];
    __syncthreads();
#pragma unroll
    for (int rr = 0; rr < 32; rr += 8)
        Wt[(size_t)(n0 + r0 + rr) * Kd + k0 + c] = t[c][r0 + rr];
}

// LayerNorm + modulate -> bf16
__global__ __launch_bounds__(256) void ln_mod(const float* __restrict__ x,
                                              const float* __restrict__ sh,
                                              const float* __restrict__ sc,
                                              bf16* __restrict__ out) {
    __shared__ float s1[256], s2[256];
    int tid = threadIdx.x;
    const float* xr = x + (size_t)blockIdx.x * HID;
    float a = 0.f, b = 0.f;
    for (int c = tid; c < HID; c += 256) { float v = xr[c]; a += v; b += v * v; }
    s1[tid] = a; s2[tid] = b; __syncthreads();
    for (int s = 128; s > 0; s >>= 1) {
        if (tid < s) { s1[tid] += s1[tid + s]; s2[tid] += s2[tid + s]; }
        __syncthreads();
    }
    float mean = s1[0] * (1.f / HID);
    float var  = s2[0] * (1.f / HID) - mean * mean;
    float rstd = rsqrtf(var + EPSF);
    bf16* orow = out + (size_t)blockIdx.x * HID;
    for (int c = tid; c < HID; c += 256)
        orow[c] = (bf16)((xr[c] - mean) * rstd * (1.f + sc[c]) + sh[c]);
}

// ---------------------------------------------------------------------------
// Generic bf16 WMMA GEMM: C[M,N] = A[M,K] x Bt[N,K]^T, 128x128 block tile.
// EPI: 0 = fp32 store; 1 = bias+gelu->bf16; 2 = resid + acc*gate -> fp32;
//      3 = resid + (acc+bias)*gate -> fp32
template <int EPI>
__global__ __launch_bounds__(256) void gemm_bf16(const bf16* __restrict__ A,
                                                 const bf16* __restrict__ Bt,
                                                 float* __restrict__ C,
                                                 bf16* __restrict__ Cbf,
                                                 const float* __restrict__ bias,
                                                 const float* __restrict__ gate,
                                                 const float* __restrict__ resid,
                                                 int M, int N, int K) {
    __shared__ bf16 Al[128][40];
    __shared__ bf16 Bl[128][40];
    const int tid = threadIdx.x, lane = tid & 31, wave = tid >> 5;
    const int wm = wave >> 1, wn = wave & 1;
    const int khalf = lane >> 4, r = lane & 15;
    const size_t m0 = (size_t)blockIdx.y * 128, n0 = (size_t)blockIdx.x * 128;

    v8f acc[2][4];
#pragma unroll
    for (int i = 0; i < 2; ++i)
#pragma unroll
        for (int j = 0; j < 4; ++j) acc[i][j] = vzero8();

    const int lrow = tid >> 1, lh = (tid & 1) * 16;
    const bf16* aptr = A + (m0 + lrow) * (size_t)K + lh;
    const bf16* bptr = Bt + (n0 + lrow) * (size_t)K + lh;

    for (int kk = 0; kk < K; kk += 32) {
        *(v8bf*)&Al[lrow][lh]     = *(const v8bf*)(aptr);
        *(v8bf*)&Al[lrow][lh + 8] = *(const v8bf*)(aptr + 8);
        *(v8bf*)&Bl[lrow][lh]     = *(const v8bf*)(bptr);
        *(v8bf*)&Bl[lrow][lh + 8] = *(const v8bf*)(bptr + 8);
        aptr += 32; bptr += 32;
        __builtin_prefetch(aptr, 0, 1);   // global_prefetch next tile
        __builtin_prefetch(bptr, 0, 1);
        __syncthreads();

        v16bf af[2], bfr[4];
        af[0] = frag_ld(&Al[wm * 32 + r][0], khalf);
        af[1] = frag_ld(&Al[wm * 32 + 16 + r][0], khalf);
#pragma unroll
        for (int j = 0; j < 4; ++j)
            bfr[j] = frag_ld(&Bl[wn * 64 + j * 16 + r][0], khalf);
#pragma unroll
        for (int i = 0; i < 2; ++i)
#pragma unroll
            for (int j = 0; j < 4; ++j)
                acc[i][j] = wmma_bf16(af[i], bfr[j], acc[i][j]);
        __syncthreads();
    }

#pragma unroll
    for (int i = 0; i < 2; ++i) {
        size_t mb = m0 + wm * 32 + i * 16 + khalf * 8;
#pragma unroll
        for (int j = 0; j < 4; ++j) {
            size_t col = n0 + wn * 64 + j * 16 + r;
#pragma unroll
            for (int v = 0; v < 8; ++v) {
                size_t idx = (mb + v) * (size_t)N + col;
                float val = acc[i][j][v];
                if (EPI == 0) {
                    C[idx] = val;
                } else if (EPI == 1) {
                    float x = val + bias[col];
                    float g = 0.5f * x *
                        (1.f + tanhf(0.7978845608f * (x + 0.044715f * x * x * x)));
                    Cbf[idx] = (bf16)g;
                } else if (EPI == 2) {
                    C[idx] = resid[idx] + val * gate[col];
                } else {
                    C[idx] = resid[idx] + (val + bias[col]) * gate[col];
                }
            }
        }
    }
}

// ---------------------------------------------------------------------------
// RMS-norm q/k, RoPE on img tokens, scale q, scatter to attention layouts.
__global__ __launch_bounds__(128) void qkv_post(const float* __restrict__ qkv,
                                                const float* __restrict__ iqn,
                                                const float* __restrict__ ikn,
                                                const float* __restrict__ tqn,
                                                const float* __restrict__ tkn,
                                                const float* __restrict__ fcos,
                                                const float* __restrict__ fsin,
                                                bf16* __restrict__ qb,
                                                bf16* __restrict__ kb,
                                                bf16* __restrict__ vt) {
    __shared__ float red[128];
    const int l = blockIdx.x, h = blockIdx.y, d = threadIdx.x;
    const float* base = qkv + (size_t)l * QKVN + h * HD + d;
    float q = base[0], k = base[HID], v = base[2 * HID];

    red[d] = q * q; __syncthreads();
    for (int s = 64; s > 0; s >>= 1) { if (d < s) red[d] += red[d + s]; __syncthreads(); }
    float qss = red[0]; __syncthreads();
    red[d] = k * k; __syncthreads();
    for (int s = 64; s > 0; s >>= 1) { if (d < s) red[d] += red[d + s]; __syncthreads(); }
    float kss = red[0];

    bool is_img = (l < L_IMG);
    float qn = q * rsqrtf(qss * (1.f / HD) + EPSF) * (is_img ? iqn[d] : tqn[d]);
    float kn = k * rsqrtf(kss * (1.f / HD) + EPSF) * (is_img ? ikn[d] : tkn[d]);

    if (is_img) {
        float c = fcos[(size_t)l * HD + d], s = fsin[(size_t)l * HD + d];
        float qp = __shfl_xor(qn, 1), kp = __shfl_xor(kn, 1);
        float qr = (d & 1) ? qp : -qp;
        float kr = (d & 1) ? kp : -kp;
        qn = qn * c + qr * s;
        kn = kn * c + kr * s;
    }
    qn *= 0.08838834764831845f;   // HD^-0.5 folded into q

    size_t hl = ((size_t)h * L_TOT + l) * HD + d;
    qb[hl] = (bf16)qn;
    kb[hl] = (bf16)kn;
    vt[((size_t)h * HD + d) * L_TOT + l] = (bf16)v;   // V stored transposed
}

// ---------------------------------------------------------------------------
// Flash attention: block = (head, 128 q rows), 8 waves x 16 q rows each.
__global__ __launch_bounds__(256) void attn_kernel(const bf16* __restrict__ qbuf,
                                                   const bf16* __restrict__ kbuf,
                                                   const bf16* __restrict__ vtbuf,
                                                   bf16* __restrict__ outb) {
    __shared__ bf16 Kl[64][136];      // [key][hd]
    __shared__ bf16 Vl[HD][72];       // [hd][key]   (V^T tile)
    __shared__ bf16 Pl[8][16][72];    // per-wave P  [m][key]

    const int tid = threadIdx.x, lane = tid & 31, wave = tid >> 5;
    const int khalf = lane >> 4, r = lane & 15;
    const int h = blockIdx.y;
    const size_t q0 = (size_t)blockIdx.x * 128;
    const bf16* qh = qbuf + (size_t)h * L_TOT * HD;
    const bf16* kh = kbuf + (size_t)h * L_TOT * HD;
    const bf16* vh = vtbuf + (size_t)h * HD * L_TOT;

    v16bf qf[4];
    {
        const bf16* qrow = qh + (q0 + wave * 16 + r) * HD;
#pragma unroll
        for (int s = 0; s < 4; ++s) qf[s] = frag_ld(qrow + 32 * s, khalf);
    }

    v8f oacc[8];
    float rmax[8], rsum[8];
#pragma unroll
    for (int t = 0; t < 8; ++t) oacc[t] = vzero8();
#pragma unroll
    for (int v = 0; v < 8; ++v) { rmax[v] = -1e30f; rsum[v] = 0.f; }

    for (int kbk = 0; kbk < L_TOT; kbk += 64) {
        {   // K tile: 64 x 128
            int row = tid >> 2, p = (tid & 3) * 32;
            const bf16* src = kh + (size_t)(kbk + row) * HD + p;
#pragma unroll
            for (int u = 0; u < 4; ++u)
                *(v8bf*)&Kl[row][p + 8 * u] = *(const v8bf*)(src + 8 * u);
        }
        {   // V^T tile: 128 x 64
            int row = tid >> 1, p = (tid & 1) * 32;
            const bf16* src = vh + (size_t)row * L_TOT + kbk + p;
#pragma unroll
            for (int u = 0; u < 4; ++u)
                *(v8bf*)&Vl[row][p + 8 * u] = *(const v8bf*)(src + 8 * u);
        }
        __syncthreads();

        // S = Q K^T  (16 x 64 per wave)
        v8f sacc[4];
#pragma unroll
        for (int j = 0; j < 4; ++j) sacc[j] = vzero8();
#pragma unroll
        for (int s = 0; s < 4; ++s)
#pragma unroll
            for (int j = 0; j < 4; ++j) {
                v16bf kf = frag_ld(&Kl[j * 16 + r][32 * s], khalf);
                sacc[j] = wmma_bf16(qf[s], kf, sacc[j]);
            }

        // online softmax (row m = khalf*8 + v)
#pragma unroll
        for (int v = 0; v < 8; ++v) {
            float mv = fmaxf(fmaxf(sacc[0][v], sacc[1][v]),
                             fmaxf(sacc[2][v], sacc[3][v]));
            mv = fmaxf(mv, __shfl_xor(mv, 1));
            mv = fmaxf(mv, __shfl_xor(mv, 2));
            mv = fmaxf(mv, __shfl_xor(mv, 4));
            mv = fmaxf(mv, __shfl_xor(mv, 8));
            float nmax = fmaxf(rmax[v], mv);
            float fac = __expf(rmax[v] - nmax);
            rmax[v] = nmax;
            rsum[v] *= fac;
#pragma unroll
            for (int t = 0; t < 8; ++t) oacc[t][v] *= fac;
            float ls = 0.f;
#pragma unroll
            for (int j = 0; j < 4; ++j) {
                float p = __expf(sacc[j][v] - nmax);
                Pl[wave][khalf * 8 + v][j * 16 + r] = (bf16)p;
                ls += p;
            }
            ls += __shfl_xor(ls, 1);
            ls += __shfl_xor(ls, 2);
            ls += __shfl_xor(ls, 4);
            ls += __shfl_xor(ls, 8);
            rsum[v] += ls;
        }
        // per-wave P writes must land before our own ds_loads below
        asm volatile("s_wait_dscnt 0" ::: "memory");

        // O += P V   (P: 16x64 A-operand, V^T: [hd][key] B-operand)
#pragma unroll
        for (int s = 0; s < 2; ++s) {
            v16bf pf = frag_ld(&Pl[wave][r][32 * s], khalf);
#pragma unroll
            for (int t = 0; t < 8; ++t) {
                v16bf vf = frag_ld(&Vl[t * 16 + r][32 * s], khalf);
                oacc[t] = wmma_bf16(pf, vf, oacc[t]);
            }
        }
        __syncthreads();
    }

    // write attn [l][h*HD + d] in bf16 for the projection GEMM
#pragma unroll
    for (int t = 0; t < 8; ++t)
#pragma unroll
        for (int v = 0; v < 8; ++v) {
            size_t row = q0 + wave * 16 + khalf * 8 + v;
            size_t col = (size_t)h * HD + t * 16 + r;
            outb[row * HID + col] = (bf16)(oacc[t][v] / rsum[v]);
        }
}

// ---------------------------------------------------------------------------
extern "C" void kernel_launch(void* const* d_in, const int* in_sizes, int n_in,
                              void* d_out, int out_size, void* d_ws, size_t ws_size,
                              hipStream_t stream) {
    const float* img       = (const float*)d_in[0];
    const float* txt       = (const float*)d_in[1];
    const float* vec       = (const float*)d_in[2];
    const float* fcos      = (const float*)d_in[3];
    const float* fsin      = (const float*)d_in[4];
    const float* img_mod_w = (const float*)d_in[5];
    const float* img_mod_b = (const float*)d_in[6];
    const float* img_qkv_w = (const float*)d_in[7];
    const float* img_qn_w  = (const float*)d_in[8];
    const float* img_kn_w  = (const float*)d_in[9];
    const float* img_proj_w= (const float*)d_in[10];
    const float* img_mlp_w1= (const float*)d_in[11];
    const float* img_mlp_b1= (const float*)d_in[12];
    const float* img_mlp_w2= (const float*)d_in[13];
    const float* img_mlp_b2= (const float*)d_in[14];
    const float* txt_mod_w = (const float*)d_in[15];
    const float* txt_mod_b = (const float*)d_in[16];
    const float* txt_qkv_w = (const float*)d_in[17];
    const float* txt_qn_w  = (const float*)d_in[18];
    const float* txt_kn_w  = (const float*)d_in[19];
    const float* txt_proj_w= (const float*)d_in[20];
    const float* txt_mlp_w1= (const float*)d_in[21];
    const float* txt_mlp_b1= (const float*)d_in[22];
    const float* txt_mlp_w2= (const float*)d_in[23];
    const float* txt_mlp_b2= (const float*)d_in[24];
    float* out = (float*)d_out;

    char* w = (char*)d_ws;
    size_t off = 0;
    auto alloc = [&](size_t bytes) -> void* {
        void* p = w + off;
        off = (off + bytes + 255) & ~(size_t)255;
        return p;
    };

    float* va     = (float*)alloc((size_t)HID * 4);
    float* mod_i  = (float*)alloc((size_t)6 * HID * 4);
    float* mod_t  = (float*)alloc((size_t)6 * HID * 4);
    bf16* qkvT_i  = (bf16*)alloc((size_t)QKVN * HID * 2);
    bf16* qkvT_t  = (bf16*)alloc((size_t)QKVN * HID * 2);
    bf16* projT_i = (bf16*)alloc((size_t)HID * HID * 2);
    bf16* projT_t = (bf16*)alloc((size_t)HID * HID * 2);
    bf16* mlp1T_i = (bf16*)alloc((size_t)MLPD * HID * 2);
    bf16* mlp1T_t = (bf16*)alloc((size_t)MLPD * HID * 2);
    bf16* mlp2T_i = (bf16*)alloc((size_t)HID * MLPD * 2);
    bf16* mlp2T_t = (bf16*)alloc((size_t)HID * MLPD * 2);
    bf16* a_bf    = (bf16*)alloc((size_t)L_TOT * HID * 2);
    float* qkv_f  = (float*)alloc((size_t)L_TOT * QKVN * 4);
    bf16* qbuf    = (bf16*)alloc((size_t)HEADS * L_TOT * HD * 2);
    bf16* kbuf    = (bf16*)alloc((size_t)HEADS * L_TOT * HD * 2);
    bf16* vtbuf   = (bf16*)alloc((size_t)HEADS * HD * L_TOT * 2);
    bf16* attn_bf = (bf16*)alloc((size_t)L_TOT * HID * 2);
    float* x2     = (float*)alloc((size_t)L_TOT * HID * 4);
    bf16* h_bf    = (bf16*)alloc((size_t)L_TOT * MLPD * 2);
    (void)ws_size; (void)in_sizes; (void)n_in; (void)out_size;

    // 1. modulation
    silu_kernel<<<12, 256, 0, stream>>>(vec, va);
    mod_gemv<<<144, 256, 0, stream>>>(va, img_mod_w, img_mod_b,
                                      txt_mod_w, txt_mod_b, mod_i, mod_t);

    // 2. weight convert + transpose (fp32 [K][N] -> bf16 [N][K])
    w_to_bf16_t<<<dim3(QKVN / 32, HID / 32), 256, 0, stream>>>(img_qkv_w, qkvT_i, HID, QKVN);
    w_to_bf16_t<<<dim3(QKVN / 32, HID / 32), 256, 0, stream>>>(txt_qkv_w, qkvT_t, HID, QKVN);
    w_to_bf16_t<<<dim3(HID / 32, HID / 32), 256, 0, stream>>>(img_proj_w, projT_i, HID, HID);
    w_to_bf16_t<<<dim3(HID / 32, HID / 32), 256, 0, stream>>>(txt_proj_w, projT_t, HID, HID);
    w_to_bf16_t<<<dim3(MLPD / 32, HID / 32), 256, 0, stream>>>(img_mlp_w1, mlp1T_i, HID, MLPD);
    w_to_bf16_t<<<dim3(MLPD / 32, HID / 32), 256, 0, stream>>>(txt_mlp_w1, mlp1T_t, HID, MLPD);
    w_to_bf16_t<<<dim3(HID / 32, MLPD / 32), 256, 0, stream>>>(img_mlp_w2, mlp2T_i, MLPD, HID);
    w_to_bf16_t<<<dim3(HID / 32, MLPD / 32), 256, 0, stream>>>(txt_mlp_w2, mlp2T_t, MLPD, HID);

    // 3. LN + modulate -> bf16 activations
    ln_mod<<<L_IMG, 256, 0, stream>>>(img, mod_i, mod_i + HID, a_bf);
    ln_mod<<<L_TXT, 256, 0, stream>>>(txt, mod_t, mod_t + HID,
                                      a_bf + (size_t)L_IMG * HID);

    // 4. QKV GEMMs
    gemm_bf16<0><<<dim3(QKVN / 128, L_IMG / 128), 256, 0, stream>>>(
        a_bf, qkvT_i, qkv_f, nullptr, nullptr, nullptr, nullptr, L_IMG, QKVN, HID);
    gemm_bf16<0><<<dim3(QKVN / 128, L_TXT / 128), 256, 0, stream>>>(
        a_bf + (size_t)L_IMG * HID, qkvT_t, qkv_f + (size_t)L_IMG * QKVN,
        nullptr, nullptr, nullptr, nullptr, L_TXT, QKVN, HID);

    // 5. RMS / RoPE / layout
    qkv_post<<<dim3(L_TOT, HEADS), 128, 0, stream>>>(
        qkv_f, img_qn_w, img_kn_w, txt_qn_w, txt_kn_w, fcos, fsin, qbuf, kbuf, vtbuf);

    // 6. flash attention
    attn_kernel<<<dim3(L_TOT / 128, HEADS), 256, 0, stream>>>(qbuf, kbuf, vtbuf, attn_bf);

    // 7. proj + gated residual -> x2
    gemm_bf16<2><<<dim3(HID / 128, L_IMG / 128), 256, 0, stream>>>(
        attn_bf, projT_i, x2, nullptr, nullptr, mod_i + 2 * HID, img, L_IMG, HID, HID);
    gemm_bf16<2><<<dim3(HID / 128, L_TXT / 128), 256, 0, stream>>>(
        attn_bf + (size_t)L_IMG * HID, projT_t, x2 + (size_t)L_IMG * HID,
        nullptr, nullptr, mod_t + 2 * HID, txt, L_TXT, HID, HID);

    // 8. LN2 + modulate
    ln_mod<<<L_IMG, 256, 0, stream>>>(x2, mod_i + 3 * HID, mod_i + 4 * HID, a_bf);
    ln_mod<<<L_TXT, 256, 0, stream>>>(x2 + (size_t)L_IMG * HID, mod_t + 3 * HID,
                                      mod_t + 4 * HID, a_bf + (size_t)L_IMG * HID);

    // 9. MLP1 (bias + gelu -> bf16)
    gemm_bf16<1><<<dim3(MLPD / 128, L_IMG / 128), 256, 0, stream>>>(
        a_bf, mlp1T_i, nullptr, h_bf, img_mlp_b1, nullptr, nullptr, L_IMG, MLPD, HID);
    gemm_bf16<1><<<dim3(MLPD / 128, L_TXT / 128), 256, 0, stream>>>(
        a_bf + (size_t)L_IMG * HID, mlp1T_t, nullptr, h_bf + (size_t)L_IMG * MLPD,
        txt_mlp_b1, nullptr, nullptr, L_TXT, MLPD, HID);

    // 10. MLP2 (bias, gate, residual) -> d_out
    gemm_bf16<3><<<dim3(HID / 128, L_IMG / 128), 256, 0, stream>>>(
        h_bf, mlp2T_i, out, nullptr, img_mlp_b2, mod_i + 5 * HID, x2, L_IMG, HID, MLPD);
    gemm_bf16<3><<<dim3(HID / 128, L_TXT / 128), 256, 0, stream>>>(
        h_bf + (size_t)L_IMG * MLPD, mlp2T_t, out + (size_t)L_IMG * HID, nullptr,
        txt_mlp_b2, mod_t + 5 * HID, x2 + (size_t)L_IMG * HID, L_TXT, HID, MLPD);
}
